// MultiLayerHGNN_66211215835667
// MI455X (gfx1250) — matrix operand order repaired and checked by reading, not compiled, not run on hardware
//
#include <hip/hip_runtime.h>
#include <hip/hip_bf16.h>
#include <stdint.h>

#define NV 100000
#define NE 20000
#define NP 800000
#define DIM 128
#define NLAYERS 3

typedef __attribute__((ext_vector_type(16))) __bf16 v16bf;
typedef __attribute__((ext_vector_type(8)))  __bf16 v8bf;
typedef __attribute__((ext_vector_type(8)))  float  v8f;

// ---------- helpers ----------
static __device__ __forceinline__ unsigned short f2bf(float f) {
    union { float f; unsigned u; } c; c.f = f;
    unsigned u = c.u;
    unsigned r = (u + 0x7FFFu + ((u >> 16) & 1u)) >> 16;  // round-to-nearest-even
    return (unsigned short)r;
}

// ---------- per-pair counting: edge degree, vertex degree, per-vertex max ----------
__global__ __launch_bounds__(256)
void k_pair_count(const int* __restrict__ pe, const int* __restrict__ pv,
                  const float* __restrict__ ew,
                  int* __restrict__ deg_e, int* __restrict__ deg_v,
                  int* __restrict__ vmax_i, int n) {
    int p = blockIdx.x * blockDim.x + threadIdx.x;
    if (p >= n) return;
    atomicAdd(deg_e + pe[p], 1);
    atomicAdd(deg_v + pv[p], 1);
    // e2v_weight is uniform [0,1) >= 0, so int-ordered max == float max
    atomicMax(vmax_i + pv[p], __float_as_int(ew[p]));
}

// ---------- exp(w - max) and per-vertex sum ----------
__global__ __launch_bounds__(256)
void k_pair_expsum(const int* __restrict__ pv, const float* __restrict__ ew,
                   const int* __restrict__ vmax_i,
                   float* __restrict__ w_e2v, float* __restrict__ vsum, int n) {
    int p = blockIdx.x * blockDim.x + threadIdx.x;
    if (p >= n) return;
    int v = pv[p];
    float ex = expf(ew[p] - __int_as_float(vmax_i[v]));
    w_e2v[p] = ex;
    atomicAdd(vsum + v, ex);
}

// ---------- normalize both pair-weight arrays ----------
__global__ __launch_bounds__(256)
void k_pair_norm(const int* __restrict__ pe, const int* __restrict__ pv,
                 const int* __restrict__ deg_e, const float* __restrict__ vsum,
                 float* __restrict__ w_v2e, float* __restrict__ w_e2v, int n) {
    int p = blockIdx.x * blockDim.x + threadIdx.x;
    if (p >= n) return;
    w_e2v[p] = w_e2v[p] / vsum[pv[p]];
    w_v2e[p] = 1.0f / (float)deg_e[pe[p]];  // softmax of all-ones == 1/count
}

// ---------- single-block hierarchical exclusive scan: off[0..n], off[n]=total ----------
__global__ __launch_bounds__(1024)
void k_scan(const int* __restrict__ cnt, int* __restrict__ off, int n) {
    __shared__ int sh[1024];
    __shared__ int carry;
    const int tid = threadIdx.x;
    if (tid == 0) carry = 0;
    __syncthreads();
    for (int base = 0; base < n; base += 1024) {
        int i = base + tid;
        int x = (i < n) ? cnt[i] : 0;
        sh[tid] = x;
        __syncthreads();
        // Hillis-Steele inclusive scan
        for (int d = 1; d < 1024; d <<= 1) {
            int t = (tid >= d) ? sh[tid - d] : 0;
            __syncthreads();
            sh[tid] += t;
            __syncthreads();
        }
        int incl = sh[tid];
        int c = carry;
        if (i < n) off[i] = c + incl - x;   // exclusive
        __syncthreads();
        if (tid == 1023) carry = c + incl;  // chunk total (OOB lanes contributed 0)
        __syncthreads();
    }
    if (tid == 0) off[n] = carry;
}

// ---------- CSR fill: bucket pair ids per edge and per vertex ----------
__global__ __launch_bounds__(256)
void k_fill(const int* __restrict__ pe, const int* __restrict__ pv,
            int* __restrict__ e_cur, int* __restrict__ v_cur,
            int* __restrict__ e_pairs, int* __restrict__ v_pairs, int n) {
    int p = blockIdx.x * blockDim.x + threadIdx.x;
    if (p >= n) return;
    e_pairs[atomicAdd(e_cur + pe[p], 1)] = p;
    v_pairs[atomicAdd(v_cur + pv[p], 1)] = p;
}

// ---------- W[l][k][n] (f32) -> Wbf[l][n][k] (bf16, column-contiguous) ----------
__global__ __launch_bounds__(256)
void k_conv_w(const float* __restrict__ Ws, unsigned short* __restrict__ Wbf, int n) {
    int i = blockIdx.x * blockDim.x + threadIdx.x;
    if (i >= n) return;
    int l   = i / (DIM * DIM);
    int rem = i - l * (DIM * DIM);
    int k   = rem / DIM;
    int nn  = rem - k * DIM;
    Wbf[l * DIM * DIM + nn * DIM + k] = f2bf(Ws[i]);
}

// ---------- X (f32) -> Xbf (bf16), 4 elements per thread ----------
__global__ __launch_bounds__(256)
void k_conv_x(const float* __restrict__ X, unsigned short* __restrict__ Xbf, int n4) {
    int i = blockIdx.x * blockDim.x + threadIdx.x;
    if (i >= n4) return;
    float4 v = reinterpret_cast<const float4*>(X)[i];
    ushort4 o;
    o.x = f2bf(v.x); o.y = f2bf(v.y); o.z = f2bf(v.z); o.w = f2bf(v.w);
    reinterpret_cast<ushort4*>(Xbf)[i] = o;
}

// ---------- WMMA GEMM: Y[NV x 128] = Xbf @ W(l) + b(l) ----------
// One wave owns a 16-row stripe and sweeps all 8 N-tiles. min-waves=1 lets the
// allocator keep all 4 A fragments and a full k-step of 8 B fragments live so
// loads clause together and WMMAs issue back-to-back.
__global__ __launch_bounds__(256, 1)
void k_gemm_wmma(const unsigned short* __restrict__ Xbf,
                 const unsigned short* __restrict__ WbfL,   // [n][k] bf16
                 const float* __restrict__ bias,
                 float* __restrict__ Y) {
    const int lane = threadIdx.x & 31;
    const int wid  = threadIdx.x >> 5;
    const int lm   = lane & 15;
    const int hi   = lane >> 4;
    const int m0   = (blockIdx.x * 8 + wid) * 16;
    if (m0 >= NV) return;

    // A fragment base: row (m0+lm), K-offset hi*16 (contiguous, matches B)
    const unsigned short* Arow  = Xbf  + (size_t)(m0 + lm) * DIM + hi * 16;
    // B fragment base: column lm of tile t lives at row (t*16+lm) of [n][k] layout
    const unsigned short* Bbase = WbfL + (size_t)lm * DIM + hi * 16;

    // Hoist all 4 A fragments (one per 32-wide k-step)
    v16bf afrag[4];
#pragma unroll
    for (int k = 0; k < 4; ++k) {
        v8bf alo = *reinterpret_cast<const v8bf*>(Arow + k * 32);       // K+0..7
        v8bf ahi = *reinterpret_cast<const v8bf*>(Arow + k * 32 + 8);   // K+8..15
        afrag[k] = __builtin_shufflevector(alo, ahi,
                      0,1,2,3,4,5,6,7,8,9,10,11,12,13,14,15);
    }

    // 8 accumulators, initialized with bias (D[m][n] += b[n])
    v8f acc[8];
#pragma unroll
    for (int t = 0; t < 8; ++t) {
        float bv = bias[t * 16 + lm];
#pragma unroll
        for (int v = 0; v < 8; ++v) acc[t][v] = bv;
    }

#pragma unroll
    for (int k = 0; k < 4; ++k) {
        // Batch-load this k-step's 8 B fragments (16 x b128 -> one clause)
        v16bf bfrag[8];
#pragma unroll
        for (int t = 0; t < 8; ++t) {
            bfrag[t] = *reinterpret_cast<const v16bf*>(
                           Bbase + (size_t)t * 16 * DIM + k * 32);
        }
#pragma unroll
        for (int t = 0; t < 8; ++t) {
            acc[t] = __builtin_amdgcn_wmma_f32_16x16x32_bf16(
                        false, afrag[k], false, bfrag[t], (short)0, acc[t],
                        false, false);
        }
    }

    // D layout: VGPR v, lanes 0-15 -> (M=v, N=lane); lanes 16-31 -> (M=v+8, N=lane-16)
#pragma unroll
    for (int t = 0; t < 8; ++t) {
#pragma unroll
        for (int v = 0; v < 8; ++v) {
            Y[(size_t)(m0 + v + hi * 8) * DIM + t * 16 + lm] = acc[t][v];
        }
    }
}

// ---------- atomic-free segmented reduction ----------
// dst[s] = sum_{p in segment s} w[p] * src[idx[p]]   (one wave32 per segment,
// float4 per lane over the 128-dim row; optional fused ReLU).
__global__ __launch_bounds__(256)
void k_gather_seg(const float* __restrict__ src, float* __restrict__ dst,
                  const int* __restrict__ off, const int* __restrict__ pairs,
                  const int* __restrict__ idx, const float* __restrict__ w,
                  int nseg, int do_relu) {
    int s = blockIdx.x * (blockDim.x >> 5) + (threadIdx.x >> 5);
    if (s >= nseg) return;
    int lane = threadIdx.x & 31;
    int b = off[s];
    int e = off[s + 1];
    float4 acc = make_float4(0.f, 0.f, 0.f, 0.f);
    for (int j = b; j < e; ++j) {
        int p = pairs[j];
        float ww = w[p];
        const float4 val = reinterpret_cast<const float4*>(
                               src + (size_t)idx[p] * DIM)[lane];
        acc.x += ww * val.x;
        acc.y += ww * val.y;
        acc.z += ww * val.z;
        acc.w += ww * val.w;
    }
    if (do_relu) {
        acc.x = fmaxf(acc.x, 0.f);
        acc.y = fmaxf(acc.y, 0.f);
        acc.z = fmaxf(acc.z, 0.f);
        acc.w = fmaxf(acc.w, 0.f);
    }
    reinterpret_cast<float4*>(dst + (size_t)s * DIM)[lane] = acc;
}

// ==================================================================
extern "C" void kernel_launch(void* const* d_in, const int* in_sizes, int n_in,
                              void* d_out, int out_size, void* d_ws, size_t ws_size,
                              hipStream_t stream) {
    const float* X0   = (const float*)d_in[0];   // [NV, DIM]
    const float* Ws   = (const float*)d_in[1];   // [3, DIM, DIM]
    const float* bs   = (const float*)d_in[2];   // [3, DIM]
    const float* e2vw = (const float*)d_in[3];   // [NP]
    const int*   pv   = (const int*)d_in[4];     // [NP]
    const int*   pe   = (const int*)d_in[5];     // [NP]
    float*       out  = (float*)d_out;           // [NV, DIM]

    // ---- workspace carve-up (256B aligned) ----
    char* wp = (char*)d_ws;
    auto carve = [&](size_t bytes) -> void* {
        void* r = (void*)wp;
        wp += (bytes + 255) & ~(size_t)255;
        return r;
    };
    float*          Y       = (float*)carve((size_t)NV * DIM * 4);   // GEMM output
    float*          Xping   = (float*)carve((size_t)NV * DIM * 4);   // layer ping buffer
    float*          Xe      = (float*)carve((size_t)NE * DIM * 4);   // edge features
    unsigned short* Xbf     = (unsigned short*)carve((size_t)NV * DIM * 2);
    unsigned short* Wbf     = (unsigned short*)carve((size_t)NLAYERS * DIM * DIM * 2);
    float*          w_v2e   = (float*)carve((size_t)NP * 4);
    float*          w_e2v   = (float*)carve((size_t)NP * 4);
    int*            deg_e   = (int*)carve((size_t)NE * 4);
    int*            deg_v   = (int*)carve((size_t)NV * 4);
    int*            vmaxi   = (int*)carve((size_t)NV * 4);
    float*          vsum    = (float*)carve((size_t)NV * 4);
    int*            e_off   = (int*)carve((size_t)(NE + 1) * 4);
    int*            v_off   = (int*)carve((size_t)(NV + 1) * 4);
    int*            e_cur   = (int*)carve((size_t)NE * 4);
    int*            v_cur   = (int*)carve((size_t)NV * 4);
    int*            e_pairs = (int*)carve((size_t)NP * 4);
    int*            v_pairs = (int*)carve((size_t)NP * 4);

    const int TPB = 256;
    const int gP  = (NP + TPB - 1) / TPB;

    // ---- pair-weight + CSR precompute (layer-invariant) ----
    hipMemsetAsync(deg_e, 0, (size_t)NE * 4, stream);
    hipMemsetAsync(deg_v, 0, (size_t)NV * 4, stream);
    hipMemsetAsync(vmaxi, 0, (size_t)NV * 4, stream);   // 0.0f bits; weights >= 0
    hipMemsetAsync(vsum,  0, (size_t)NV * 4, stream);
    k_pair_count <<<gP, TPB, 0, stream>>>(pe, pv, e2vw, deg_e, deg_v, vmaxi, NP);
    k_pair_expsum<<<gP, TPB, 0, stream>>>(pv, e2vw, vmaxi, w_e2v, vsum, NP);
    k_pair_norm  <<<gP, TPB, 0, stream>>>(pe, pv, deg_e, vsum, w_v2e, w_e2v, NP);

    k_scan<<<1, 1024, 0, stream>>>(deg_e, e_off, NE);
    k_scan<<<1, 1024, 0, stream>>>(deg_v, v_off, NV);
    hipMemcpyAsync(e_cur, e_off, (size_t)NE * 4, hipMemcpyDeviceToDevice, stream);
    hipMemcpyAsync(v_cur, v_off, (size_t)NV * 4, hipMemcpyDeviceToDevice, stream);
    k_fill<<<gP, TPB, 0, stream>>>(pe, pv, e_cur, v_cur, e_pairs, v_pairs, NP);

    // ---- weight matrices -> bf16, transposed to [n][k] ----
    {
        int n = NLAYERS * DIM * DIM;
        k_conv_w<<<(n + TPB - 1) / TPB, TPB, 0, stream>>>(Ws, Wbf, n);
    }

    // ---- layers ----
    const int n4     = NV * DIM / 4;
    const int gConv  = (n4 + TPB - 1) / TPB;
    const int gGemm  = (NV / 16 + 7) / 8;       // 8 waves x 16-row stripes per block
    const int gSegE  = (NE + 7) / 8;            // 8 segments (waves) per block
    const int gSegV  = (NV + 7) / 8;

    const float* Xcur = X0;
    for (int l = 0; l < NLAYERS; ++l) {
        // theta: Y = Xcur @ W[l] + b[l]  (bf16 WMMA, f32 accumulate)
        k_conv_x<<<gConv, TPB, 0, stream>>>(Xcur, Xbf, n4);
        k_gemm_wmma<<<gGemm, TPB, 0, stream>>>(Xbf,
                                               Wbf + (size_t)l * DIM * DIM,
                                               bs + (size_t)l * DIM,
                                               Y);

        // v2e: Xe[e] = sum_p w_v2e[p] * Y[pv[p]]   (atomic-free)
        k_gather_seg<<<gSegE, TPB, 0, stream>>>(Y, Xe, e_off, e_pairs, pv,
                                                w_v2e, NE, /*relu=*/0);

        // e2v: Xnext[v] = relu(sum_p w_e2v[p] * Xe[pe[p]])
        float* Xnext = (l == NLAYERS - 1) ? out : Xping;
        k_gather_seg<<<gSegV, TPB, 0, stream>>>(Xe, Xnext, v_off, v_pairs, pe,
                                                w_e2v, NV, /*relu=*/1);

        Xcur = Xnext;
    }
}
